// LocalCCMambaBlock_19129784336933
// MI455X (gfx1250) — matrix operand order repaired and checked by reading, not compiled, not run on hardware
//
#include <hip/hip_runtime.h>
#include <hip/hip_bf16.h>

typedef __bf16 bf16;
typedef __attribute__((ext_vector_type(16))) __bf16 v16bf;
typedef __attribute__((ext_vector_type(8)))  __bf16 bf16x8;
typedef __attribute__((ext_vector_type(8)))  float  v8f;

#define NSEQ    4096
#define DMODEL  256
#define DSTATE  16
#define DCONV   4
#define DINNER  512
#define DTRANK  16
#define NDBC    48          // DTRANK + 2*DSTATE
#define EPS     1e-5f

// ---------------------------------------------------------------------------
// f32 -> bf16 elementwise convert (grid-stride)
// ---------------------------------------------------------------------------
__global__ void cvt_bf16(const float* __restrict__ src, bf16* __restrict__ dst, int n) {
    int i = blockIdx.x * blockDim.x + threadIdx.x;
    int stride = gridDim.x * blockDim.x;
    for (; i < n; i += stride) dst[i] = (bf16)src[i];
}

// transpose + convert: src (R x C) f32 row-major -> dst (C x R) bf16 row-major
__global__ void cvtT_bf16(const float* __restrict__ src, bf16* __restrict__ dst,
                          int R, int C) {
    int i = blockIdx.x * blockDim.x + threadIdx.x;
    int n = R * C;
    int stride = gridDim.x * blockDim.x;
    for (; i < n; i += stride) {
        int r = i / C, c = i % C;
        dst[(size_t)c * R + r] = (bf16)src[i];
    }
}

// ---------------------------------------------------------------------------
// A-fragment loader: 16x32 bf16 tile, row m = lane%16.
// lanes 0-15:  K = kb + {0..15};  lanes 16-31: K = kb+16 + {0..15}
// (kb already includes + lh*16). Two contiguous 16B chunks per lane.
// ---------------------------------------------------------------------------
__device__ __forceinline__ v16bf load_a_frag(const bf16* __restrict__ p) {
    bf16x8 c0 = *(const bf16x8*)(p);
    bf16x8 c1 = *(const bf16x8*)(p + 8);
    v16bf a;
#pragma unroll
    for (int i = 0; i < 8; ++i) { a[i] = c0[i]; a[8 + i] = c1[i]; }
    return a;
}

// B-fragment loader from Bt (row = output column): lane holds N=lo,
// K = k0 + lh*16 + {0..15} -> one contiguous 32B run of Bt.
__device__ __forceinline__ v16bf load_b_frag(const bf16* __restrict__ p) {
    bf16x8 c0 = *(const bf16x8*)(p);
    bf16x8 c1 = *(const bf16x8*)(p + 8);
    v16bf b;
#pragma unroll
    for (int i = 0; i < 8; ++i) { b[i] = c0[i]; b[8 + i] = c1[i]; }
    return b;
}

// ---------------------------------------------------------------------------
// bf16 WMMA GEMM:  C[M,N] (f32) = A[M,K] (bf16 rm) * B,  B given as Bt[N,K].
// Block = 256 threads = 8 waves. Wave computes a 32 x (JT*16) tile:
//   2 M-subtiles x JT N-subtiles = 2*JT WMMAs per K-step, branch-free body.
// Requires: M % 256 == 0, K % 32 == 0, N >= blockIdx.y*JT*16 + JT*16.
// ---------------------------------------------------------------------------
template <int JT>
__global__ __launch_bounds__(256) void wmma_gemm_bf16(
    const bf16* __restrict__ A, const bf16* __restrict__ Bt,
    float* __restrict__ C, int M, int N, int K) {
    const int wave = threadIdx.x >> 5;
    const int lane = threadIdx.x & 31;
    const int lh   = lane >> 4;      // 0: lanes 0-15, 1: lanes 16-31
    const int lo   = lane & 15;
    const int m0   = (blockIdx.x * 8 + wave) * 32;
    const int n0   = blockIdx.y * (JT * 16);

    v8f acc[2][JT];
#pragma unroll
    for (int mi = 0; mi < 2; ++mi)
#pragma unroll
        for (int j = 0; j < JT; ++j)
#pragma unroll
            for (int i = 0; i < 8; ++i) acc[mi][j][i] = 0.0f;

    const bf16* __restrict__ Arow0 = A + (size_t)(m0 + lo) * K;
    const bf16* __restrict__ Arow1 = A + (size_t)(m0 + 16 + lo) * K;
    const bf16* __restrict__ Bbase = Bt + (size_t)(n0 + lo) * K;

    for (int k0 = 0; k0 < K; k0 += 32) {
        const int kb = k0 + lh * 16;
        const v16bf a0 = load_a_frag(Arow0 + kb);
        const v16bf a1 = load_a_frag(Arow1 + kb);
        __builtin_prefetch(Arow0 + kb + 256, 0, 0);   // global_prefetch_b8
        __builtin_prefetch(Arow1 + kb + 256, 0, 0);

        v16bf b[JT];
#pragma unroll
        for (int j = 0; j < JT; ++j)
            b[j] = load_b_frag(Bbase + (size_t)j * 16 * K + k0 + lh * 16);

#pragma unroll
        for (int j = 0; j < JT; ++j)
            acc[0][j] = __builtin_amdgcn_wmma_f32_16x16x32_bf16(
                false, a0, false, b[j], (short)0, acc[0][j], false, false);
#pragma unroll
        for (int j = 0; j < JT; ++j)
            acc[1][j] = __builtin_amdgcn_wmma_f32_16x16x32_bf16(
                false, a1, false, b[j], (short)0, acc[1][j], false, false);
    }

    // store: VGPR r of subtile (mi,j) -> row m0 + mi*16 + r + lh*8, col n0 + j*16 + lo
#pragma unroll
    for (int mi = 0; mi < 2; ++mi)
#pragma unroll
        for (int j = 0; j < JT; ++j)
#pragma unroll
            for (int r = 0; r < 8; ++r)
                C[(size_t)(m0 + mi * 16 + r + lh * 8) * N + n0 + j * 16 + lo] =
                    acc[mi][j][r];
}

// ---------------------------------------------------------------------------
// bias + ReLU + LayerNorm over rows of hpre (NSEQ x DMODEL); bf16 out.
// ---------------------------------------------------------------------------
__global__ __launch_bounds__(256) void bias_relu_ln(
    const float* __restrict__ hpre, const float* __restrict__ gcn_b,
    const float* __restrict__ ln_g, const float* __restrict__ ln_b,
    bf16* __restrict__ hbf) {
    __shared__ float ssum[256];
    __shared__ float ssq[256];
    const int row = blockIdx.x, tid = threadIdx.x;
    float v = hpre[(size_t)row * DMODEL + tid] + gcn_b[tid];
    v = fmaxf(v, 0.0f);
    ssum[tid] = v;
    ssq[tid]  = v * v;
    __syncthreads();
    for (int s = 128; s > 0; s >>= 1) {
        if (tid < s) { ssum[tid] += ssum[tid + s]; ssq[tid] += ssq[tid + s]; }
        __syncthreads();
    }
    const float mu  = ssum[0] * (1.0f / DMODEL);
    const float var = ssq[0] * (1.0f / DMODEL) - mu * mu;
    const float o = (v - mu) * rsqrtf(var + EPS) * ln_g[tid] + ln_b[tid];
    hbf[(size_t)row * DMODEL + tid] = (bf16)o;
}

// ---------------------------------------------------------------------------
// causal depthwise conv (width 4) + bias + SiLU. Outputs u (f32) and u_bf16.
// ---------------------------------------------------------------------------
__global__ __launch_bounds__(256) void conv_silu(
    const float* __restrict__ xz, const float* __restrict__ conv_w,
    const float* __restrict__ conv_b,
    float* __restrict__ u, bf16* __restrict__ ubf) {
    const int idx = blockIdx.x * blockDim.x + threadIdx.x;
    if (idx >= NSEQ * DINNER) return;
    const int t = idx / DINNER, c = idx % DINNER;
    float acc = conv_b[c];
#pragma unroll
    for (int k = 0; k < DCONV; ++k) {
        const int tt = t + k - (DCONV - 1);
        if (tt >= 0)
            acc += xz[(size_t)tt * (2 * DINNER) + c] * conv_w[c * DCONV + k];
    }
    const float s = acc / (1.0f + __expf(-acc));
    u[idx]   = s;
    ubf[idx] = (bf16)s;
}

// ---------------------------------------------------------------------------
// dt = softplus(dbc[:, :DTRANK] @ w_dt + b_dt)     (K = 16 -> plain VALU)
// ---------------------------------------------------------------------------
__global__ __launch_bounds__(256) void dt_softplus(
    const float* __restrict__ dbc, const float* __restrict__ w_dt,
    const float* __restrict__ b_dt, float* __restrict__ dt) {
    const int idx = blockIdx.x * blockDim.x + threadIdx.x;
    if (idx >= NSEQ * DINNER) return;
    const int t = idx / DINNER, c = idx % DINNER;
    float acc = b_dt[c];
#pragma unroll
    for (int r = 0; r < DTRANK; ++r)
        acc += dbc[t * NDBC + r] * w_dt[r * DINNER + c];
    dt[idx] = (acc > 20.0f) ? acc : log1pf(__expf(acc));
}

// ---------------------------------------------------------------------------
// Async global->LDS copy of one f32 (CDNA5 GLOBAL_LOAD_ASYNC_TO_LDS_B32,
// tracked by ASYNCcnt). dst is an LDS byte offset, src a global address.
// ---------------------------------------------------------------------------
__device__ __forceinline__ void async_copy_f32(unsigned lds_off,
                                               const float* __restrict__ gptr) {
    asm volatile("global_load_async_to_lds_b32 %0, %1, off"
                 :
                 : "v"(lds_off), "v"(gptr)
                 : "memory");
}

__device__ __forceinline__ void async_wait0() {
    asm volatile("s_wait_asynccnt 0" ::: "memory");
}

// ---------------------------------------------------------------------------
// Sequential SSM scan, LDS-staged in chunks of 128 timesteps via async
// global->LDS loads. 32 blocks x 256 threads; block owns 16 channels;
// thread = (channel, state). Fused: y = (h@C + u*D) * silu(z) -> bf16.
// ---------------------------------------------------------------------------
#define SCAN_T 128
__global__ __launch_bounds__(256) void ssm_scan(
    const float* __restrict__ dt, const float* __restrict__ u,
    const float* __restrict__ dbc, const float* __restrict__ A_log,
    const float* __restrict__ D_skip, const float* __restrict__ xz,
    bf16* __restrict__ ybf) {
    __shared__ float dt_s[SCAN_T][16];
    __shared__ float u_s[SCAN_T][16];
    __shared__ float B_s[SCAN_T][DSTATE];
    __shared__ float C_s[SCAN_T][DSTATE];
    const int tid   = threadIdx.x;
    const int s     = tid & 15;        // state index
    const int cl    = tid >> 4;        // channel within block, 0..15
    const int cbase = blockIdx.x * 16;
    const int c     = cbase + cl;
    const float Acs = -__expf(A_log[c * DSTATE + s]);
    const float Dc  = D_skip[c];
    float hst = 0.0f;

    for (int tb = 0; tb < NSEQ; tb += SCAN_T) {
        // async-stage this chunk's dt/u/B/C straight into LDS
        for (int i = tid; i < SCAN_T * 16; i += 256) {
            const int tt = i >> 4, cc = i & 15;
            async_copy_f32((unsigned)(uintptr_t)&dt_s[tt][cc],
                           dt + (size_t)(tb + tt) * DINNER + cbase + cc);
            async_copy_f32((unsigned)(uintptr_t)&u_s[tt][cc],
                           u + (size_t)(tb + tt) * DINNER + cbase + cc);
            async_copy_f32((unsigned)(uintptr_t)&B_s[tt][cc],
                           dbc + (size_t)(tb + tt) * NDBC + DTRANK + cc);
            async_copy_f32((unsigned)(uintptr_t)&C_s[tt][cc],
                           dbc + (size_t)(tb + tt) * NDBC + DTRANK + DSTATE + cc);
        }
        async_wait0();
        __syncthreads();

        for (int tt = 0; tt < SCAN_T; ++tt) {
            const float dtv = dt_s[tt][cl];
            const float uv  = u_s[tt][cl];
            hst = __expf(dtv * Acs) * hst + dtv * B_s[tt][s] * uv;
            float p = hst * C_s[tt][s];
            p += __shfl_xor(p, 8, 16);
            p += __shfl_xor(p, 4, 16);
            p += __shfl_xor(p, 2, 16);
            p += __shfl_xor(p, 1, 16);
            if (s == 0) {
                const int t  = tb + tt;
                const float zv = xz[(size_t)t * (2 * DINNER) + DINNER + c];
                const float yv = (p + uv * Dc) * (zv / (1.0f + __expf(-zv)));
                ybf[(size_t)t * DINNER + c] = (bf16)yv;
            }
        }
        __syncthreads();
    }
}

// ---------------------------------------------------------------------------
extern "C" void kernel_launch(void* const* d_in, const int* in_sizes, int n_in,
                              void* d_out, int out_size, void* d_ws, size_t ws_size,
                              hipStream_t stream) {
    const float* x      = (const float*)d_in[0];
    const float* adj    = (const float*)d_in[1];
    const float* gcn_w  = (const float*)d_in[2];
    const float* gcn_b  = (const float*)d_in[3];
    const float* ln_g   = (const float*)d_in[4];
    const float* ln_b   = (const float*)d_in[5];
    const float* w_in   = (const float*)d_in[6];
    const float* conv_w = (const float*)d_in[7];
    const float* conv_b = (const float*)d_in[8];
    const float* w_xp   = (const float*)d_in[9];
    const float* w_dt   = (const float*)d_in[10];
    const float* b_dt   = (const float*)d_in[11];
    const float* A_log  = (const float*)d_in[12];
    const float* D_skip = (const float*)d_in[13];
    const float* w_out  = (const float*)d_in[14];
    float* out = (float*)d_out;

    char* ws = (char*)d_ws;
    size_t off = 0;
    auto carve = [&](size_t bytes) -> char* {
        char* p = ws + off;
        off += (bytes + 255) & ~(size_t)255;
        return p;
    };
    bf16* x_bf   = (bf16*)carve(sizeof(bf16) * NSEQ * DMODEL);
    bf16* gcnwT  = (bf16*)carve(sizeof(bf16) * DMODEL * DMODEL);
    float* xg    = (float*)carve(sizeof(float) * NSEQ * DMODEL);
    bf16* adj_bf = (bf16*)carve(sizeof(bf16) * (size_t)NSEQ * NSEQ);
    bf16* xgT    = (bf16*)carve(sizeof(bf16) * DMODEL * NSEQ);
    float* hpre  = (float*)carve(sizeof(float) * NSEQ * DMODEL);
    bf16* h_bf   = (bf16*)carve(sizeof(bf16) * NSEQ * DMODEL);
    bf16* winT   = (bf16*)carve(sizeof(bf16) * (2 * DINNER) * DMODEL);
    float* xz    = (float*)carve(sizeof(float) * NSEQ * (2 * DINNER));
    float* u     = (float*)carve(sizeof(float) * NSEQ * DINNER);
    bf16* u_bf   = (bf16*)carve(sizeof(bf16) * NSEQ * DINNER);
    bf16* wxpT   = (bf16*)carve(sizeof(bf16) * NDBC * DINNER);
    float* dbc   = (float*)carve(sizeof(float) * NSEQ * NDBC);
    float* dt    = (float*)carve(sizeof(float) * NSEQ * DINNER);
    bf16* y_bf   = (bf16*)carve(sizeof(bf16) * NSEQ * DINNER);
    bf16* woutT  = (bf16*)carve(sizeof(bf16) * DMODEL * DINNER);
    (void)ws_size; (void)in_sizes; (void)n_in; (void)out_size;

    const dim3 blk(256);
    const int gx = NSEQ / 256;   // 16 blocks in M (8 waves x 32 rows each)

    // 1) xg = x @ gcn_w  (bf16 WMMA)
    cvt_bf16<<<2048, blk, 0, stream>>>(x, x_bf, NSEQ * DMODEL);
    cvtT_bf16<<<256, blk, 0, stream>>>(gcn_w, gcnwT, DMODEL, DMODEL);
    wmma_gemm_bf16<4><<<dim3(gx, DMODEL / 64), blk, 0, stream>>>(
        x_bf, gcnwT, xg, NSEQ, DMODEL, DMODEL);

    // 2) hpre = adj @ xg   (the big GEMM, 8.6 GFLOP)
    cvt_bf16<<<4096, blk, 0, stream>>>(adj, adj_bf, NSEQ * NSEQ);
    cvtT_bf16<<<2048, blk, 0, stream>>>(xg, xgT, NSEQ, DMODEL);
    wmma_gemm_bf16<4><<<dim3(gx, DMODEL / 64), blk, 0, stream>>>(
        adj_bf, xgT, hpre, NSEQ, DMODEL, NSEQ);

    // 3) h = LN(relu(hpre + b))
    bias_relu_ln<<<NSEQ, blk, 0, stream>>>(hpre, gcn_b, ln_g, ln_b, h_bf);

    // 4) xz = h @ w_in
    cvtT_bf16<<<1024, blk, 0, stream>>>(w_in, winT, DMODEL, 2 * DINNER);
    wmma_gemm_bf16<4><<<dim3(gx, (2 * DINNER) / 64), blk, 0, stream>>>(
        h_bf, winT, xz, NSEQ, 2 * DINNER, DMODEL);

    // 5) u = silu(conv(xc) + b)
    conv_silu<<<(NSEQ * DINNER) / 256, blk, 0, stream>>>(xz, conv_w, conv_b, u, u_bf);

    // 6) dbc = u @ w_xp   (N = 48 -> 3 subtiles, branch-free)
    cvtT_bf16<<<128, blk, 0, stream>>>(w_xp, wxpT, DINNER, NDBC);
    wmma_gemm_bf16<3><<<dim3(gx, 1), blk, 0, stream>>>(
        u_bf, wxpT, dbc, NSEQ, NDBC, DINNER);

    // 7) dt = softplus(dbc[:, :16] @ w_dt + b_dt)
    dt_softplus<<<(NSEQ * DINNER) / 256, blk, 0, stream>>>(dbc, w_dt, b_dt, dt);

    // 8) selective scan + fused gating -> y (bf16), async LDS staging
    ssm_scan<<<DINNER / 16, blk, 0, stream>>>(dt, u, dbc, A_log, D_skip, xz, y_bf);

    // 9) out = y @ w_out  (f32 straight into d_out)
    cvtT_bf16<<<512, blk, 0, stream>>>(w_out, woutT, DINNER, DMODEL);
    wmma_gemm_bf16<4><<<dim3(gx, DMODEL / 64), blk, 0, stream>>>(
        y_bf, woutT, out, NSEQ, DMODEL, DINNER);
}
